// GATLayer_88613765251205
// MI455X (gfx1250) — matrix operand order repaired and checked by reading, not compiled, not run on hardware
//
#include <hip/hip_runtime.h>

typedef __attribute__((ext_vector_type(16))) _Float16 v16h;
typedef __attribute__((ext_vector_type(8)))  float    v8f;
typedef __attribute__((ext_vector_type(4)))  int      v4i;

#define GN    8192
#define GFIN  128
#define GFOUT 64
#define GALPHA 0.2f
#define LOG2E 1.4426950408889634f

// ---------------------------------------------------------------------------
// Kernel 1: h = X @ W  (f32 accumulate), write ht (f16, transposed [n][j]),
//           s1 = (h@a1)*log2e, s2 = (h@a2)*log2e.
// (log2e pre-scale is exact through leakyrelu since lrelu is positively
//  homogeneous; lets the attention kernel use native exp2.)
// Block = 256 threads handles 16 rows of X. Grid = 512.
// ---------------------------------------------------------------------------
__global__ void gat_prep(const float* __restrict__ X, const float* __restrict__ W,
                         const float* __restrict__ a1, const float* __restrict__ a2,
                         _Float16* __restrict__ ht, float* __restrict__ s1,
                         float* __restrict__ s2) {
    __shared__ float Xs[16][GFIN];
    __shared__ float hs[16][GFOUT];
    const int tid  = threadIdx.x;
    const int row0 = blockIdx.x * 16;

    for (int i = tid; i < 16 * GFIN; i += 256)
        Xs[i / GFIN][i % GFIN] = X[(size_t)(row0 + i / GFIN) * GFIN + (i % GFIN)];
    __syncthreads();

    #pragma unroll
    for (int q = 0; q < 4; ++q) {
        int idx = tid + 256 * q;           // 0..1023
        int r = idx >> 6, c = idx & 63;
        float acc = 0.f;
        #pragma unroll 8
        for (int k = 0; k < GFIN; ++k)
            acc += Xs[r][k] * W[k * GFOUT + c];
        hs[r][c] = acc;
        ht[(size_t)c * GN + row0 + r] = (_Float16)acc;
    }
    __syncthreads();

    if (tid < 16) {
        float v1 = 0.f, v2 = 0.f;
        #pragma unroll 8
        for (int c = 0; c < GFOUT; ++c) {
            float h = hs[tid][c];
            v1 += h * a1[c];
            v2 += h * a2[c];
        }
        s1[row0 + tid] = v1 * LOG2E;
        s2[row0 + tid] = v2 * LOG2E;
    }
}

// ---------------------------------------------------------------------------
// Kernel 2: s2max = max_j s2[j]   (single block)
// ---------------------------------------------------------------------------
__global__ void gat_s2max(const float* __restrict__ s2, float* __restrict__ s2mx) {
    __shared__ float red[32];
    float m = -3.4e38f;
    for (int i = threadIdx.x; i < GN; i += 1024)
        m = fmaxf(m, s2[i]);
    #pragma unroll
    for (int off = 16; off > 0; off >>= 1)
        m = fmaxf(m, __shfl_down(m, off, 32));
    if ((threadIdx.x & 31) == 0) red[threadIdx.x >> 5] = m;
    __syncthreads();
    if (threadIdx.x < 32) {
        m = red[threadIdx.x];
        #pragma unroll
        for (int off = 16; off > 0; off >>= 1)
            m = fmaxf(m, __shfl_down(m, off, 32));
        if (threadIdx.x == 0) *s2mx = m;
    }
}

// ---------------------------------------------------------------------------
// Kernel 3: fused masked-softmax-attention row pass.
// Block = 256 threads (8 waves), owns 16 output rows. Each wave streams an
// interleaved set of 32-wide column tiles: builds P (f16, WMMA-A layout)
// from A-mask + exp2(lrelu(s1_i+s2_j) - shift_i), accumulates P @ h via
// 4x v_wmma_f32_16x16x32_f16 N-slices + a 5th WMMA against an all-ones B
// that produces the softmax row-sums on the matrix pipe (no serial VALU
// reduction chain). Block-reduces in LDS, writes out = acc / rowsum.
// ---------------------------------------------------------------------------
__global__ void __launch_bounds__(256)
gat_attn(const int* __restrict__ Adj, const float* __restrict__ s1,
         const float* __restrict__ s2, const float* __restrict__ s2maxp,
         const _Float16* __restrict__ ht, float* __restrict__ out) {
    __shared__ float lds_rs[16];
    __shared__ float lds_acc[16][GFOUT];

    const int tid  = threadIdx.x;
    const int lane = tid & 31;
    const int wave = tid >> 5;
    const int row0 = blockIdx.x * 16;
    const int m    = lane & 15;     // row within tile / N within B fragment
    const int hi   = lane >> 4;     // half-wave select

    // zero LDS reduction buffers
    if (tid < 16) lds_rs[tid] = 0.f;
    #pragma unroll
    for (int q = 0; q < 4; ++q) {
        int idx = tid + 256 * q;
        lds_acc[idx >> 6][idx & 63] = 0.f;
    }
    __syncthreads();

    const float s2max = *s2maxp;
    const float s1m   = s1[row0 + m];
    const float xsh   = s1m + s2max;
    const float shift = fmaxf(xsh, GALPHA * xsh);   // lrelu(s1_i + max s2), log2 units

    // K offsets for this lane (contiguous order, matching B):
    // lane<16: K in {0..15}; lane>=16: K in {16..31}
    const int kb0 = hi * 16;
    const int kb1 = hi * 16 + 8;

    v8f acc0 = {}, acc1 = {}, acc2 = {}, acc3 = {}, accS = {};
    v16h ones;
    #pragma unroll
    for (int i = 0; i < 16; ++i) ones[i] = (_Float16)1.0f;

    const int* Arow = Adj + (size_t)(row0 + m) * GN;

    for (int t = wave; t < GN / 32; t += 8) {
        const int col0 = t * 32;

        // adjacency mask: 16 ints, four 128-bit loads, streamed (non-temporal)
        const int* ap = Arow + col0;
        v4i A0 = __builtin_nontemporal_load((const v4i*)(ap + kb0));
        v4i A1 = __builtin_nontemporal_load((const v4i*)(ap + kb0 + 4));
        v4i A2 = __builtin_nontemporal_load((const v4i*)(ap + kb1));
        v4i A3 = __builtin_nontemporal_load((const v4i*)(ap + kb1 + 4));

        // s2 values for this lane's K set (hot in cache)
        float4 Q0 = *(const float4*)(s2 + col0 + kb0);
        float4 Q1 = *(const float4*)(s2 + col0 + kb0 + 4);
        float4 Q2 = *(const float4*)(s2 + col0 + kb1);
        float4 Q3 = *(const float4*)(s2 + col0 + kb1 + 4);

        auto pv = [&](int a, float s2k) -> _Float16 {
            float x = s1m + s2k;
            float e = fmaxf(x, GALPHA * x) - shift;          // <= 0 always
            float p = a ? __builtin_amdgcn_exp2f(e) : 0.f;   // native v_exp_f32
            return (_Float16)p;
        };

        v16h P;
        P[0]  = pv(A0.x, Q0.x); P[1]  = pv(A0.y, Q0.y);
        P[2]  = pv(A0.z, Q0.z); P[3]  = pv(A0.w, Q0.w);
        P[4]  = pv(A1.x, Q1.x); P[5]  = pv(A1.y, Q1.y);
        P[6]  = pv(A1.z, Q1.z); P[7]  = pv(A1.w, Q1.w);
        P[8]  = pv(A2.x, Q2.x); P[9]  = pv(A2.y, Q2.y);
        P[10] = pv(A2.z, Q2.z); P[11] = pv(A2.w, Q2.w);
        P[12] = pv(A3.x, Q3.x); P[13] = pv(A3.y, Q3.y);
        P[14] = pv(A3.z, Q3.z); P[15] = pv(A3.w, Q3.w);

        // B operand: ht[n][j] transposed f16 -> contiguous 32B per lane.
        // b[i] holds h[col0 + i + 16*hi][n0 + m]
        const _Float16* hb = ht + col0 + 16 * hi;
        v16h B0 = *(const v16h*)(hb + (size_t)( 0 + m) * GN);
        v16h B1 = *(const v16h*)(hb + (size_t)(16 + m) * GN);
        v16h B2 = *(const v16h*)(hb + (size_t)(32 + m) * GN);
        v16h B3 = *(const v16h*)(hb + (size_t)(48 + m) * GN);

        acc0 = __builtin_amdgcn_wmma_f32_16x16x32_f16(false, P, false, B0,
                                                      (short)0, acc0, false, false);
        acc1 = __builtin_amdgcn_wmma_f32_16x16x32_f16(false, P, false, B1,
                                                      (short)0, acc1, false, false);
        acc2 = __builtin_amdgcn_wmma_f32_16x16x32_f16(false, P, false, B2,
                                                      (short)0, acc2, false, false);
        acc3 = __builtin_amdgcn_wmma_f32_16x16x32_f16(false, P, false, B3,
                                                      (short)0, acc3, false, false);
        // row-sums on the matrix pipe: D[m][*] = sum_k P[m][k]
        accS = __builtin_amdgcn_wmma_f32_16x16x32_f16(false, P, false, ones,
                                                      (short)0, accS, false, false);
    }

    // block-wide reduction: row sums (only N==0 lanes carry a unique copy)
    if (m == 0) {
        #pragma unroll
        for (int r = 0; r < 8; ++r)
            atomicAdd(&lds_rs[r + 8 * hi], accS[r]);
    }
    #pragma unroll
    for (int r = 0; r < 8; ++r) {
        int row = r + 8 * hi;   // C/D layout: VGPR r -> M = r + 8*(lane>=16)
        atomicAdd(&lds_acc[row][ 0 + m], acc0[r]);
        atomicAdd(&lds_acc[row][16 + m], acc1[r]);
        atomicAdd(&lds_acc[row][32 + m], acc2[r]);
        atomicAdd(&lds_acc[row][48 + m], acc3[r]);
    }
    __syncthreads();

    #pragma unroll
    for (int q = 0; q < 4; ++q) {
        int idx = tid + 256 * q;
        int r = idx >> 6, c = idx & 63;
        out[(size_t)(row0 + r) * GFOUT + c] = lds_acc[r][c] / lds_rs[r];
    }
}

// ---------------------------------------------------------------------------
extern "C" void kernel_launch(void* const* d_in, const int* in_sizes, int n_in,
                              void* d_out, int out_size, void* d_ws, size_t ws_size,
                              hipStream_t stream) {
    const float* X  = (const float*)d_in[0];
    const int*   A  = (const int*)d_in[1];
    const float* W  = (const float*)d_in[2];
    const float* a1 = (const float*)d_in[3];
    const float* a2 = (const float*)d_in[4];
    float* out = (float*)d_out;

    char* ws = (char*)d_ws;
    _Float16* ht = (_Float16*)ws;                             // 64*8192*2 = 1 MB
    float* s1    = (float*)(ws + (size_t)GFOUT * GN * 2);     // 32 KB
    float* s2    = s1 + GN;                                   // 32 KB
    float* s2mx  = s2 + GN;                                   // 4 B

    gat_prep <<<GN / 16, 256, 0, stream>>>(X, W, a1, a2, ht, s1, s2);
    gat_s2max<<<1, 1024, 0, stream>>>(s2, s2mx);
    gat_attn <<<GN / 16, 256, 0, stream>>>(A, s1, s2, s2mx, ht, out);
}
